// MD_LSTM_35124242547241
// MI455X (gfx1250) — matrix operand order repaired and checked by reading, not compiled, not run on hardware
//
#include <hip/hip_runtime.h>
#include <hip/hip_bf16.h>

// Problem constants (reference: B, L, IN, H, T = 512, 16, 128, 1024, 64)
#define B_   512
#define L_   16
#define IN_  128
#define H_   1024
#define T_   64
#define G4H  (4 * H_)        // 4096 gate columns
#define K0_  (IN_ + H_)      // 1152  (layer0 concat K)
#define K1_  (2 * H_)        // 2048  (layer1 concat K)
#define SEQ_ (L_ + T_)       // 80 output time slices

typedef __attribute__((ext_vector_type(16))) __bf16 v16bf;
typedef __attribute__((ext_vector_type(8)))  __bf16 v8bf;
typedef __attribute__((ext_vector_type(8)))  float  v8f;

// ---------------- precompute / utility kernels ----------------

// dst[r, 0:Ka] = bf16(a[r, :]), dst[r, Ka:Ka+Kb] = bf16(b[r, :])
__global__ void pack_wcat(__bf16* __restrict__ dst,
                          const float* __restrict__ a, int Ka,
                          const float* __restrict__ b, int Kb, int rows) {
    long long idx = (long long)blockIdx.x * 256 + threadIdx.x;
    int K = Ka + Kb;
    long long total = (long long)rows * K;
    if (idx >= total) return;
    int r = (int)(idx / K), c = (int)(idx % K);
    float v = (c < Ka) ? a[(long long)r * Ka + c] : b[(long long)r * Kb + (c - Ka)];
    dst[idx] = (__bf16)v;
}

__global__ void cvt_f32_to_bf16(__bf16* __restrict__ dst, const float* __restrict__ src, int n) {
    int i = blockIdx.x * 256 + threadIdx.x;
    if (i < n) dst[i] = (__bf16)src[i];
}

__global__ void bias_sum(float* __restrict__ dst, const float* __restrict__ a,
                         const float* __restrict__ b, int n) {
    int i = blockIdx.x * 256 + threadIdx.x;
    if (i < n) dst[i] = a[i] + b[i];
}

__global__ void zero_f32(float* p, int n) {
    int i = blockIdx.x * 256 + threadIdx.x;
    if (i < n) p[i] = 0.0f;
}
__global__ void zero_bf16(__bf16* p, int n) {
    int i = blockIdx.x * 256 + threadIdx.x;
    if (i < n) p[i] = (__bf16)0.0f;
}

// out[:, 0:L, :] = x
__global__ void copy_prompt(float* __restrict__ out, const float* __restrict__ x) {
    int i = blockIdx.x * 256 + threadIdx.x;
    if (i >= B_ * L_ * IN_) return;
    int b = i / (L_ * IN_);
    int rem = i - b * (L_ * IN_);
    out[(long long)b * SEQ_ * IN_ + rem] = x[i];
}

// A0[:, 0:IN] = bf16(x[:, s, :])
__global__ void cvt_x_step(__bf16* __restrict__ A0, const float* __restrict__ x, int s) {
    int i = blockIdx.x * 256 + threadIdx.x;
    if (i >= B_ * IN_) return;
    int b = i >> 7, c = i & (IN_ - 1);
    A0[(long long)b * K0_ + c] = (__bf16)x[(long long)b * L_ * IN_ + (long long)s * IN_ + c];
}

// ---------------- bf16 WMMA GEMM: Gout[M,N] = A[M,K](lda) * W[N,K]^T ----------------
// grid = (N/64, M/128), block = 128 (4 waves); wave w computes a 32x64 tile:
// rows [128*by + 32w, +32) x cols [64*bx, +64). Register double-buffered K loop.

__device__ __forceinline__ v16bf load_a_frag(const __bf16* __restrict__ p, int k) {
    // p already includes the lane-half K offset (kha); contiguous A layout:
    // V0-7 = K [kha, kha+16)
    union { v16bf v; v8bf h[2]; } t;
    t.h[0] = *reinterpret_cast<const v8bf*>(p + k);
    t.h[1] = *reinterpret_cast<const v8bf*>(p + k + 8);
    return t.v;
}
__device__ __forceinline__ v16bf load_b_frag(const __bf16* __restrict__ p, int k) {
    // p already includes lane-half K offset (khb); 16 contiguous K values.
    return *reinterpret_cast<const v16bf*>(p + k);
}

#define WMMA_BF16(ACC, AF, BF) \
    (ACC) = __builtin_amdgcn_wmma_f32_16x16x32_bf16(false, (AF), false, (BF), (short)0, (ACC), false, false)

__global__ __launch_bounds__(128)
void gemm_bf16_wmma(const __bf16* __restrict__ A, int lda,
                    const __bf16* __restrict__ W,
                    float* __restrict__ Gout, int N, int K) {
    const int lane = threadIdx.x & 31;
    const int wave = threadIdx.x >> 5;
    const int l16  = lane & 15;
    const int hi   = lane >> 4;           // lane half
    const int tileN = blockIdx.x * 64;
    const int tileM = blockIdx.y * 128 + wave * 32;
    const int kha = hi * 16;              // A lane-half K offset
    const int khb = hi * 16;              // B lane-half K offset

    const __bf16* Ar0 = A + (long long)(tileM + l16) * lda + kha;
    const __bf16* Ar1 = Ar0 + (long long)16 * lda;
    const __bf16* Wr0 = W + (long long)(tileN + l16) * K + khb;
    const __bf16* Wr1 = Wr0 + (long long)16 * K;
    const __bf16* Wr2 = Wr0 + (long long)32 * K;
    const __bf16* Wr3 = Wr0 + (long long)48 * K;

    v8f acc00 = {}, acc01 = {}, acc02 = {}, acc03 = {};
    v8f acc10 = {}, acc11 = {}, acc12 = {}, acc13 = {};

    // prologue: fragments for k = 0
    v16bf a0 = load_a_frag(Ar0, 0), a1 = load_a_frag(Ar1, 0);
    v16bf b0 = load_b_frag(Wr0, 0), b1 = load_b_frag(Wr1, 0);
    v16bf b2 = load_b_frag(Wr2, 0), b3 = load_b_frag(Wr3, 0);

    for (int k = 32; k < K; k += 32) {
        if (k + 32 < K) __builtin_prefetch(Ar0 + k + 32, 0, 3);   // global_prefetch
        // issue next tile's loads first so WMMAs below overlap the VMEM latency
        v16bf na0 = load_a_frag(Ar0, k), na1 = load_a_frag(Ar1, k);
        v16bf nb0 = load_b_frag(Wr0, k), nb1 = load_b_frag(Wr1, k);
        v16bf nb2 = load_b_frag(Wr2, k), nb3 = load_b_frag(Wr3, k);

        WMMA_BF16(acc00, a0, b0); WMMA_BF16(acc01, a0, b1);
        WMMA_BF16(acc02, a0, b2); WMMA_BF16(acc03, a0, b3);
        WMMA_BF16(acc10, a1, b0); WMMA_BF16(acc11, a1, b1);
        WMMA_BF16(acc12, a1, b2); WMMA_BF16(acc13, a1, b3);

        a0 = na0; a1 = na1; b0 = nb0; b1 = nb1; b2 = nb2; b3 = nb3;
    }
    // final K tile
    WMMA_BF16(acc00, a0, b0); WMMA_BF16(acc01, a0, b1);
    WMMA_BF16(acc02, a0, b2); WMMA_BF16(acc03, a0, b3);
    WMMA_BF16(acc10, a1, b0); WMMA_BF16(acc11, a1, b1);
    WMMA_BF16(acc12, a1, b2); WMMA_BF16(acc13, a1, b3);

    // C/D layout: VGPR v -> row (tileM + 16*sub + hi*8 + v), col tileN + nt*16 + l16
    float* g0 = Gout + (long long)(tileM + hi * 8) * N + tileN + l16;
    float* g1 = g0 + (long long)16 * N;
#pragma unroll
    for (int v = 0; v < 8; ++v) {
        g0[(long long)v * N +  0] = acc00[v];
        g0[(long long)v * N + 16] = acc01[v];
        g0[(long long)v * N + 32] = acc02[v];
        g0[(long long)v * N + 48] = acc03[v];
        g1[(long long)v * N +  0] = acc10[v];
        g1[(long long)v * N + 16] = acc11[v];
        g1[(long long)v * N + 32] = acc12[v];
        g1[(long long)v * N + 48] = acc13[v];
    }
}

// ---------------- LSTM cell epilogue ----------------
// gates G[b, 0:4H] order i,f,g,o ; bsum = b_ih + b_hh ; writes h (bf16) to up to 2 dests
__global__ void lstm_cell(const float* __restrict__ G, const float* __restrict__ bsum,
                          float* __restrict__ c,
                          __bf16* __restrict__ hA, int ldA,
                          __bf16* __restrict__ hB, int ldB) {
    int i = blockIdx.x * 256 + threadIdx.x;
    if (i >= B_ * H_) return;
    int b = i >> 10, j = i & (H_ - 1);
    const float* g = G + (long long)b * G4H;
    float gi = g[j]          + bsum[j];
    float gf = g[j + H_]     + bsum[j + H_];
    float gg = g[j + 2*H_]   + bsum[j + 2*H_];
    float go = g[j + 3*H_]   + bsum[j + 3*H_];
    float si = 1.0f / (1.0f + __expf(-gi));
    float sf = 1.0f / (1.0f + __expf(-gf));
    float so = 1.0f / (1.0f + __expf(-go));
    float cn = sf * c[i] + si * tanhf(gg);
    c[i] = cn;
    float h = so * tanhf(cn);
    hA[(long long)b * ldA + j] = (__bf16)h;
    if (hB) hB[(long long)b * ldB + j] = (__bf16)h;
}

// ---------------- projection epilogue ----------------
// xnew = xprev + Gp + b_out ; store to out[:, slice, :], xlast, and bf16 into A0[:,0:IN]
__global__ void proj_add(const float* __restrict__ Gp, const float* __restrict__ bout,
                         const float* __restrict__ xprev, int xstride,
                         float* __restrict__ xlast, float* __restrict__ out, int slice,
                         __bf16* __restrict__ A0) {
    int i = blockIdx.x * 256 + threadIdx.x;
    if (i >= B_ * IN_) return;
    int b = i >> 7, c = i & (IN_ - 1);
    float v = xprev[(long long)b * xstride + c] + Gp[i] + bout[c];
    xlast[i] = v;
    out[(long long)b * SEQ_ * IN_ + (long long)slice * IN_ + c] = v;
    A0[(long long)b * K0_ + c] = (__bf16)v;
}

// ---------------- host orchestration ----------------

static inline int nblk(long long n) { return (int)((n + 255) / 256); }

extern "C" void kernel_launch(void* const* d_in, const int* in_sizes, int n_in,
                              void* d_out, int out_size, void* d_ws, size_t ws_size,
                              hipStream_t stream) {
    (void)in_sizes; (void)n_in; (void)out_size; (void)ws_size;
    const float* x     = (const float*)d_in[0];
    const float* W_ih0 = (const float*)d_in[1];
    const float* W_hh0 = (const float*)d_in[2];
    const float* b_ih0 = (const float*)d_in[3];
    const float* b_hh0 = (const float*)d_in[4];
    const float* W_ih1 = (const float*)d_in[5];
    const float* W_hh1 = (const float*)d_in[6];
    const float* b_ih1 = (const float*)d_in[7];
    const float* b_hh1 = (const float*)d_in[8];
    const float* W_out = (const float*)d_in[9];
    const float* b_out = (const float*)d_in[10];
    float* out = (float*)d_out;

    // workspace carve-up (256B aligned)
    char* ws = (char*)d_ws;
    size_t off = 0;
    auto carve = [&](size_t bytes) -> void* {
        void* p = ws + off;
        off = (off + bytes + 255) & ~(size_t)255;
        return p;
    };
    __bf16* Wcat0 = (__bf16*)carve((size_t)G4H * K0_ * 2);   // 9.4 MB
    __bf16* Wcat1 = (__bf16*)carve((size_t)G4H * K1_ * 2);   // 16.8 MB
    __bf16* Woutb = (__bf16*)carve((size_t)IN_ * H_ * 2);
    float*  bsum0 = (float*)carve((size_t)G4H * 4);
    float*  bsum1 = (float*)carve((size_t)G4H * 4);
    __bf16* A0    = (__bf16*)carve((size_t)B_ * K0_ * 2);    // [x | h0] bf16
    __bf16* A1    = (__bf16*)carve((size_t)B_ * K1_ * 2);    // [h0 | h1] bf16
    float*  G     = (float*)carve((size_t)B_ * G4H * 4);     // gate pre-activations
    float*  Gp    = (float*)carve((size_t)B_ * IN_ * 4);     // projection result
    float*  c0    = (float*)carve((size_t)B_ * H_ * 4);
    float*  c1    = (float*)carve((size_t)B_ * H_ * 4);
    float*  xlast = (float*)carve((size_t)B_ * IN_ * 4);

    // --- one-time (per launch) precompute ---
    pack_wcat<<<nblk((long long)G4H * K0_), 256, 0, stream>>>(Wcat0, W_ih0, IN_, W_hh0, H_, G4H);
    pack_wcat<<<nblk((long long)G4H * K1_), 256, 0, stream>>>(Wcat1, W_ih1, H_, W_hh1, H_, G4H);
    cvt_f32_to_bf16<<<nblk(IN_ * H_), 256, 0, stream>>>(Woutb, W_out, IN_ * H_);
    bias_sum<<<nblk(G4H), 256, 0, stream>>>(bsum0, b_ih0, b_hh0, G4H);
    bias_sum<<<nblk(G4H), 256, 0, stream>>>(bsum1, b_ih1, b_hh1, G4H);
    zero_bf16<<<nblk((long long)B_ * K0_), 256, 0, stream>>>(A0, B_ * K0_);
    zero_bf16<<<nblk((long long)B_ * K1_), 256, 0, stream>>>(A1, B_ * K1_);
    zero_f32<<<nblk((long long)B_ * H_), 256, 0, stream>>>(c0, B_ * H_);
    zero_f32<<<nblk((long long)B_ * H_), 256, 0, stream>>>(c1, B_ * H_);
    copy_prompt<<<nblk((long long)B_ * L_ * IN_), 256, 0, stream>>>(out, x);

    const dim3 gemm_blk(128);
    const dim3 gates_grid(G4H / 64, B_ / 128);  // (64, 4)
    const dim3 proj_grid(IN_ / 64, B_ / 128);   // (2, 4)
    const int  cell_blocks = nblk((long long)B_ * H_);
    const int  vec_blocks  = nblk((long long)B_ * IN_);

    // 79 sequential stack steps: s=0..15 prompt, s=16..78 generation
    for (int s = 0; s < L_ + T_ - 1; ++s) {
        if (s < L_)
            cvt_x_step<<<vec_blocks, 256, 0, stream>>>(A0, x, s);

        // layer 0: G = [x|h0] @ [W_ih0|W_hh0]^T
        gemm_bf16_wmma<<<gates_grid, gemm_blk, 0, stream>>>(A0, K0_, Wcat0, G, G4H, K0_);
        lstm_cell<<<cell_blocks, 256, 0, stream>>>(G, bsum0, c0,
                                                   A0 + IN_, K0_,   // h0 -> A0[:,128:]
                                                   A1, K1_);        // h0 -> A1[:,:1024]
        // layer 1: G = [h0|h1] @ [W_ih1|W_hh1]^T
        gemm_bf16_wmma<<<gates_grid, gemm_blk, 0, stream>>>(A1, K1_, Wcat1, G, G4H, K1_);
        lstm_cell<<<cell_blocks, 256, 0, stream>>>(G, bsum1, c1,
                                                   A1 + H_, K1_,    // h1 -> A1[:,1024:]
                                                   (__bf16*)nullptr, 0);

        if (s >= L_ - 1) {
            // projection: Gp = h1 @ W_out^T ; then residual add + emit slice s+1
            gemm_bf16_wmma<<<proj_grid, gemm_blk, 0, stream>>>(A1 + H_, K1_, Woutb, Gp, IN_, H_);
            const float* xprev = (s == L_ - 1) ? (x + (L_ - 1) * IN_) : xlast;
            int xstride        = (s == L_ - 1) ? (L_ * IN_) : IN_;
            proj_add<<<vec_blocks, 256, 0, stream>>>(Gp, b_out, xprev, xstride,
                                                     xlast, out, s + 1, A0);
        }
    }
}
